// GraphAttention_12532714570460
// MI455X (gfx1250) — matrix operand order repaired and checked by reading, not compiled, not run on hardware
//
#include <hip/hip_runtime.h>
#include <hip/hip_bf16.h>
#include <math.h>

// GraphAttention, MI455X/gfx1250, wave32 + v_wmma_f32_16x16x32_bf16 + TDM.
//
// Math rewrite: scores = (Q Kt + (Q Wr) relT + Q.br) / 8  -- avoids ever
// materializing rel_h = rel @ Wr.T (saves 8.6 GFLOP + 268MB of traffic).
// bf16 inputs to the matrix engine, fp32 accumulation, fp32 softmax.
// The 268MB rel stream (the HBM-bound part, ~12us at 23.3 TB/s) is moved by
// the Tensor Data Mover into LDS, double-buffered, tracked with TENSORcnt.
// Workspace: Q,K,V,QR (4 x 4MB) + QB (64KB) + scores (32MB) ~= 51MB fp32.

typedef __attribute__((ext_vector_type(16))) __bf16 v16bf;
typedef __attribute__((ext_vector_type(8)))  float  v8f;
typedef __attribute__((ext_vector_type(4)))  unsigned int u32x4;
typedef __attribute__((ext_vector_type(8)))  int i32x8;
typedef __attribute__((ext_vector_type(4)))  int i32x4;

constexpr int Bc = 4, Sc = 512, Hc = 8, DHc = 64, RDc = 64, DMc = 512;

#define DEVFN static __device__ __forceinline__

DEVFN int lane_id() { return (int)(threadIdx.x & 31u); }

DEVFN v8f wmma_bf16(v16bf a, v16bf b, v8f c) {
  // D = A(16x32) * B(32x16) + C, f32 accumulate
  return __builtin_amdgcn_wmma_f32_16x16x32_bf16(
      /*neg_a=*/false, a, /*neg_b=*/false, b,
      /*c_mod=*/(short)0, c, /*reuse_a=*/false, /*reuse_b=*/false);
}

// ---- VGPR layout helpers (ISA 7.12.2, wave32) -------------------------------
// A 16x32 bf16: lane l -> row m=l&15; VGPR p holds K-pair
//   k = (p>=4?16:0) + (l>=16?8:0) + (p&3)*2
// B 32x16 bf16: lane l -> col n=l&15; VGPR v holds K-pair k = (v>=4?16:0)+(l>=16?8:0)+(v&3)*2
// C/D 16x16 f32: lane l -> col n=l&15; VGPR r holds row m = r + (l>=16?8:0)

// A tile from fp32 row-major [16 x ld], columns k0..k0+31 (pairs contiguous)
DEVFN v16bf ldA(const float* __restrict__ src, int ld, int k0) {
  int l = lane_id();
  int m = l & 15;
  int koff = (l >> 4) * 8;
  v16bf a;
#pragma unroll
  for (int p = 0; p < 8; ++p) {
    int k = k0 + ((p & 4) ? 16 : 0) + koff + (p & 3) * 2;
    const float* s = src + m * ld + k;
    a[2 * p]     = (__bf16)s[0];
    a[2 * p + 1] = (__bf16)s[1];
  }
  return a;
}

// B tile where logical B[k][n] = src[n*ld + k0 + k] (source is N x K row-major,
// contraction dim contiguous -> float2 per pair). Used for "X * Yt" GEMMs.
DEVFN v16bf ldBt(const float* __restrict__ src, int ld, int k0) {
  int l = lane_id();
  int n = l & 15;
  int kb = (l >> 4) * 8;
  v16bf b;
#pragma unroll
  for (int v = 0; v < 8; ++v) {
    int k = k0 + ((v & 4) ? 16 : 0) + kb + (v & 3) * 2;
    const float* s = src + n * ld + k;
    b[2 * v]     = (__bf16)s[0];
    b[2 * v + 1] = (__bf16)s[1];
  }
  return b;
}

// Same as ldBt but reading a TDM-filled LDS buffer (volatile: the compiler
// never sees the producer, the Tensor Data Mover wrote it).
DEVFN v16bf ldBt_ldsv(const volatile float* src, int ld, int k0) {
  int l = lane_id();
  int n = l & 15;
  int kb = (l >> 4) * 8;
  v16bf b;
#pragma unroll
  for (int v = 0; v < 8; ++v) {
    int k = k0 + ((v & 4) ? 16 : 0) + kb + (v & 3) * 2;
    float f0 = src[n * ld + k];
    float f1 = src[n * ld + k + 1];
    b[2 * v]     = (__bf16)f0;
    b[2 * v + 1] = (__bf16)f1;
  }
  return b;
}

// B tile where logical B[k][n] = src[(k0+k)*ld + n] (source is K x N row-major)
DEVFN v16bf ldB(const float* __restrict__ src, int ld, int k0) {
  int l = lane_id();
  int n = l & 15;
  int kb = (l >> 4) * 8;
  v16bf b;
#pragma unroll
  for (int v = 0; v < 8; ++v) {
    int k = k0 + ((v & 4) ? 16 : 0) + kb + (v & 3) * 2;
    b[2 * v]     = (__bf16)src[(size_t)k * ld + n];
    b[2 * v + 1] = (__bf16)src[(size_t)(k + 1) * ld + n];
  }
  return b;
}

// Store C/D tile to fp32 row-major [16 x ld]
DEVFN void stC(float* __restrict__ dst, int ld, const v8f& c) {
  int l = lane_id();
  int n = l & 15;
  int mb = (l >> 4) * 8;
#pragma unroll
  for (int r = 0; r < 8; ++r) dst[(size_t)(mb + r) * ld + n] = c[r];
}

// ---- Tensor Data Mover: 1D contiguous 16KB (4096 x f32) load into LDS -------
// D# per ISA cdna5 ch.8: group0 = {count, lds_addr, global_addr(57b), type=2};
// group1 = {data_size=4B, tensor_dim0=4096, tensor_dim1=1, tile_dim0=4096,
//           tensor_dim0_stride=4096}; groups 2/3 zero (<=2D).
DEVFN void tdm_load_16kb(unsigned long long gaddr, unsigned int lds_addr) {
  u32x4 g0;
  g0[0] = 1u;                                   // count=1, user descriptor
  g0[1] = lds_addr;                             // LDS byte address
  g0[2] = (unsigned int)gaddr;                  // global_addr[31:0]
  g0[3] = (unsigned int)((gaddr >> 32) & 0x01FFFFFFu) | (2u << 30);  // type=2
  i32x8 g1;
  g1[0] = (int)(2u << 16);                      // data_size = 2 (4 bytes)
  g1[1] = (int)(4096u << 16);                   // tensor_dim0[15:0] = 4096
  g1[2] = (int)(1u << 16);                      // dim0 hi = 0; tensor_dim1 = 1
  g1[3] = (int)(4096u << 16);                   // dim1 hi = 0; tile_dim0 = 4096
  g1[4] = 0;                                    // tile_dim1/2 unused
  g1[5] = 4096;                                 // tensor_dim0_stride lo
  g1[6] = 0;
  g1[7] = 0;
  i32x4 z4 = {0, 0, 0, 0};
#if __clang_major__ >= 23
  i32x8 z8 = {0, 0, 0, 0, 0, 0, 0, 0};
  __builtin_amdgcn_tensor_load_to_lds(g0, g1, z4, z4, z8, 0);
#else
  __builtin_amdgcn_tensor_load_to_lds(g0, g1, z4, z4, 0);
#endif
}

// ---- kernel 1: QKV projections  out[b,h,s,d] = (x @ W.T) --------------------
__global__ __launch_bounds__(32)
void proj_kernel(const float* __restrict__ x, const float* __restrict__ W,
                 float* __restrict__ out) {
  int b  = blockIdx.z;
  int q0 = blockIdx.y * 16;
  int o0 = blockIdx.x * 16;
  const float* xa = x + ((size_t)b * Sc + q0) * DMc;
  v8f c = {};
  for (int kc = 0; kc < DMc; kc += 32) {
    v16bf a  = ldA(xa, DMc, kc);
    v16bf bm = ldBt(W + (size_t)o0 * DMc, DMc, kc);  // B[c][o] = W[o][c]
    c = wmma_bf16(a, bm, c);
  }
  int l = lane_id();
  int n = l & 15;
  int mb = (l >> 4) * 8;
  int o = o0 + n;
  int h = o >> 6, d = o & 63;  // o0 multiple of 16 => h uniform across tile rows
#pragma unroll
  for (int r = 0; r < 8; ++r) {
    int q = q0 + mb + r;
    out[(((size_t)b * Hc + h) * Sc + q) * DHc + d] = c[r];
  }
}

// ---- kernel 2: QR = Q @ Wr  (qr[b,q,h,r]),  qb[b,q,h] = Q . br --------------
__global__ __launch_bounds__(32)
void qr_kernel(const float* __restrict__ Qf, const float* __restrict__ Wr,
               const float* __restrict__ br, float* __restrict__ qr,
               float* __restrict__ qb) {
  int b  = blockIdx.z;
  int h  = blockIdx.y;
  int q0 = blockIdx.x * 16;
  const float* Qa = Qf + (((size_t)b * Hc + h) * Sc + q0) * DHc;
  v16bf a0 = ldA(Qa, DHc, 0);
  v16bf a1 = ldA(Qa, DHc, 32);
  int l = lane_id();
  int n = l & 15;
  int mb = (l >> 4) * 8;
#pragma unroll
  for (int rt = 0; rt < RDc; rt += 16) {
    v8f c = {};
    c = wmma_bf16(a0, ldB(Wr + rt, RDc, 0), c);   // B[d][r] = Wr[d][rt+n]
    c = wmma_bf16(a1, ldB(Wr + rt, RDc, 32), c);
#pragma unroll
    for (int r = 0; r < 8; ++r) {
      int q = q0 + mb + r;
      qr[(((size_t)b * Sc + q) * Hc + h) * RDc + rt + n] = c[r];
    }
  }
  if (l < 16) {  // q . br (softmax-invariant but kept for score fidelity)
    int q = q0 + l;
    const float* Qrow = Qf + (((size_t)b * Hc + h) * Sc + q) * DHc;
    float s = 0.f;
    for (int d = 0; d < DHc; ++d) s += Qrow[d] * br[d];
    qb[((size_t)b * Sc + q) * Hc + h] = s;
  }
}

// ---- kernel 3: scores = Q Kt per (b,h) --------------------------------------
__global__ __launch_bounds__(32)
void scores1_kernel(const float* __restrict__ Qf, const float* __restrict__ Kf,
                    float* __restrict__ sc) {
  int bh = blockIdx.x;
  int q0 = blockIdx.y * 16;
  const float* Qa = Qf + ((size_t)bh * Sc + q0) * DHc;
  v16bf a0 = ldA(Qa, DHc, 0);
  v16bf a1 = ldA(Qa, DHc, 32);
  const float* Kb = Kf + (size_t)bh * Sc * DHc;
  float* scp = sc + ((size_t)bh * Sc + q0) * Sc;
  for (int kt = 0; kt < Sc; kt += 16) {
    v8f c = {};
    c = wmma_bf16(a0, ldBt(Kb + (size_t)kt * DHc, DHc, 0), c);
    c = wmma_bf16(a1, ldBt(Kb + (size_t)kt * DHc, DHc, 32), c);
    stC(scp + kt, Sc, c);
  }
}

// ---- kernel 4: scores[b,h,q,:] += qr[b,q,h,:] . rel[b,q,:,:]T + qb ----------
// Per (b,q): M = 8 heads padded to 16, N = k (512), contraction = r (64).
// Streams the contiguous 128KB rel slice via the Tensor Data Mover into a
// 2 x 16KB LDS double buffer (TENSORcnt pipelined, max 2 outstanding).
__global__ __launch_bounds__(32)
void scores2_kernel(const float* __restrict__ qr, const float* __restrict__ qb,
                    const float* __restrict__ rel, float* __restrict__ sc) {
  __shared__ float relbuf[2][64 * RDc];  // 2 x 16KB = 64 k-rows per chunk
  int q = blockIdx.x;
  int b = blockIdx.y;
  int l = lane_id();

  // kick off the DMA pipeline first (uniform addresses -> SGPR descriptors)
  const unsigned long long relbase =
      (unsigned long long)(size_t)(rel + (((size_t)b * Sc + q) * Sc) * RDc);
  unsigned int lds0 = (unsigned int)(size_t)&relbuf[0][0];
  unsigned int lds1 = (unsigned int)(size_t)&relbuf[1][0];
  tdm_load_16kb(relbase, lds0);
  tdm_load_16kb(relbase + 16384ull, lds1);

  int m = l & 15;
  int koff = (l >> 4) * 8;
  const float* qra = qr + ((size_t)b * Sc + q) * Hc * RDc;
  v16bf a0, a1;  // rows 8..15 zero-padded
#pragma unroll
  for (int p = 0; p < 8; ++p) {
    int r0 = ((p & 4) ? 16 : 0) + koff + (p & 3) * 2;
    float f00 = 0.f, f01 = 0.f, f10 = 0.f, f11 = 0.f;
    if (m < Hc) {
      const float* qrow = qra + (size_t)m * RDc;
      f00 = qrow[r0];      f01 = qrow[r0 + 1];
      f10 = qrow[32 + r0]; f11 = qrow[32 + r0 + 1];
    }
    a0[2 * p] = (__bf16)f00; a0[2 * p + 1] = (__bf16)f01;
    a1[2 * p] = (__bf16)f10; a1[2 * p + 1] = (__bf16)f11;
  }
  float qbv[8];
#pragma unroll
  for (int rr = 0; rr < 8; ++rr) qbv[rr] = qb[((size_t)b * Sc + q) * Hc + rr];

  int n = l & 15;
  for (int ch = 0; ch < 8; ++ch) {       // 8 chunks x 64 k-rows
    if (ch < 7) __builtin_amdgcn_s_wait_tensorcnt(1);  // chunk ch landed
    else        __builtin_amdgcn_s_wait_tensorcnt(0);
    asm volatile("" ::: "memory");
    const volatile float* bufc = relbuf[ch & 1];
#pragma unroll
    for (int kt2 = 0; kt2 < 4; ++kt2) {
      int kt = ch * 64 + kt2 * 16;
      v8f c = {};
      if (l < 16) {  // C rows 0..7 = existing Q.Kt scores for heads 0..7
#pragma unroll
        for (int rr = 0; rr < 8; ++rr)
          c[rr] = sc[(((size_t)b * Hc + rr) * Sc + q) * Sc + kt + n];
      }
      v16bf b0 = ldBt_ldsv(bufc + kt2 * 16 * RDc, RDc, 0);   // B[r][k]=rel[k][r]
      v16bf b1 = ldBt_ldsv(bufc + kt2 * 16 * RDc, RDc, 32);
      c = wmma_bf16(a0, b0, c);          // EXEC all-1s here (straight-line)
      c = wmma_bf16(a1, b1, c);
      if (l < 16) {
#pragma unroll
        for (int rr = 0; rr < 8; ++rr)
          sc[(((size_t)b * Hc + rr) * Sc + q) * Sc + kt + n] = c[rr] + qbv[rr];
      }
    }
    if (ch + 2 < 8) {  // refill the buffer we just drained
      asm volatile("s_wait_dscnt 0x0" ::: "memory");  // LDS reads done
      tdm_load_16kb(relbase + (unsigned long long)(ch + 2) * 16384ull,
                    (ch & 1) ? lds1 : lds0);
    }
  }
}

// ---- kernel 5: mask + softmax (fp32) + P V (wmma) ---------------------------
__global__ __launch_bounds__(128)
void softmax_pv_kernel(const float* __restrict__ sc, const float* __restrict__ Vf,
                       const int* __restrict__ graph, float* __restrict__ out) {
  __shared__ float Sbuf[16][Sc];   // 32KB: masked scores, then exp()
  __shared__ float red[16][8];
  __shared__ float rowinv[16];

  int bh = blockIdx.x;
  int b = bh / Hc, h = bh % Hc;
  int q0 = blockIdx.y * 16;
  int t = (int)threadIdx.x;
  int row = t >> 3;  // 0..15
  int seg = t & 7;   // 8 threads per row, 64 k each
  int q = q0 + row;
  const float* scp = sc + ((size_t)bh * Sc + q) * Sc;
  const int*   gp  = graph + ((size_t)b * Sc + q) * Sc;

  float lm = -3.0e38f;
  for (int j = 0; j < 64; ++j) {
    int k = seg * 64 + j;
    float s = scp[k] * 0.125f + (gp[k] ? 0.0f : -1.0e9f);
    Sbuf[row][k] = s;
    lm = fmaxf(lm, s);
  }
  red[row][seg] = lm;
  __syncthreads();
  float rm = red[row][0];
#pragma unroll
  for (int j = 1; j < 8; ++j) rm = fmaxf(rm, red[row][j]);
  __syncthreads();
  float ls = 0.f;
  for (int j = 0; j < 64; ++j) {
    int k = seg * 64 + j;
    float e = __expf(Sbuf[row][k] - rm);
    Sbuf[row][k] = e;
    ls += e;
  }
  red[row][seg] = ls;
  __syncthreads();
  if (seg == 0) {
    float rs = 0.f;
#pragma unroll
    for (int j = 0; j < 8; ++j) rs += red[row][j];
    rowinv[row] = 1.0f / rs;
  }
  __syncthreads();

  // phase 2: wave w computes ctx d-tile [16w, 16w+16); normalization folded
  // into the bf16 A-tile load so accumulation stays fp32-exact per element.
  int w = t >> 5;
  int dt = w * 16;
  int l = lane_id();
  int m = l & 15;
  int koff = (l >> 4) * 8;
  float invm = rowinv[m];
  const float* Vbase = Vf + (size_t)bh * Sc * DHc + dt;
  v8f c = {};
  for (int kt = 0; kt < Sc; kt += 32) {
    v16bf a;
#pragma unroll
    for (int p = 0; p < 8; ++p) {
      int k = kt + ((p & 4) ? 16 : 0) + koff + (p & 3) * 2;
      a[2 * p]     = (__bf16)(Sbuf[m][k] * invm);
      a[2 * p + 1] = (__bf16)(Sbuf[m][k + 1] * invm);
    }
    v16bf bb = ldB(Vbase, DHc, kt);  // B[k][d] = V[k][dt+n]
    c = wmma_bf16(a, bb, c);
  }
  int n = l & 15;
  int mb = (l >> 4) * 8;
#pragma unroll
  for (int r = 0; r < 8; ++r) {
    int qq = q0 + mb + r;
    out[((size_t)b * Sc + qq) * DMc + h * DHc + dt + n] = c[r];
  }
}

extern "C" void kernel_launch(void* const* d_in, const int* in_sizes, int n_in,
                              void* d_out, int out_size, void* d_ws, size_t ws_size,
                              hipStream_t stream) {
  (void)in_sizes; (void)n_in; (void)out_size; (void)ws_size;
  const float* x   = (const float*)d_in[0];
  const int*   g   = (const int*)d_in[1];
  const float* rel = (const float*)d_in[2];
  const float* Wq  = (const float*)d_in[3];
  const float* Wk  = (const float*)d_in[4];
  const float* Wv  = (const float*)d_in[5];
  const float* Wr  = (const float*)d_in[6];
  const float* br  = (const float*)d_in[7];
  float* out = (float*)d_out;

  float* ws = (float*)d_ws;                 // needs ~51MB fp32 scratch
  const size_t NQ = (size_t)Bc * Hc * Sc * DHc;  // 1,048,576
  float* Qf = ws;
  float* Kf = Qf + NQ;
  float* Vf = Kf + NQ;
  float* QR = Vf + NQ;
  float* QB = QR + NQ;
  float* SC = QB + (size_t)Bc * Sc * Hc;    // 8,388,608 floats (32MB)

  dim3 gp(DMc / 16, Sc / 16, Bc);
  proj_kernel<<<gp, 32, 0, stream>>>(x, Wq, Qf);
  proj_kernel<<<gp, 32, 0, stream>>>(x, Wk, Kf);
  proj_kernel<<<gp, 32, 0, stream>>>(x, Wv, Vf);
  qr_kernel<<<dim3(Sc / 16, Hc, Bc), 32, 0, stream>>>(Qf, Wr, br, QR, QB);
  scores1_kernel<<<dim3(Bc * Hc, Sc / 16), 32, 0, stream>>>(Qf, Kf, SC);
  scores2_kernel<<<dim3(Sc, Bc), 32, 0, stream>>>(QR, QB, rel, SC);
  softmax_pv_kernel<<<dim3(Bc * Hc, Sc / 16), 128, 0, stream>>>(SC, Vf, g, out);
}
